// ConvBlock_42090679501105
// MI455X (gfx1250) — matrix operand order repaired and checked by reading, not compiled, not run on hardware
//
#include <hip/hip_runtime.h>

// ---------------------------------------------------------------------------
// KPConv block (KPConv -> GroupNorm -> LeakyReLU) for MI455X (gfx1250, wave32)
//
// Roofline: ~410MB random feature gather vs 9.2 GFLOP of matmul ->
// gather-bandwidth bound. Design:
//  - 16 queries per 128-thread block (4 waves), LDS ~114KB -> 2 blocks/WGP.
//  - global_prefetch of neighbor rows overlaps gather latency with the
//    VALU-heavy influence-weight phase.
//  - GEMM1 (per query): weighted(16x64) = w^T(16x32) x F(32x64) via
//    v_wmma_f32_16x16x32_f16 (K=15 padded to 16 with a zero row).
//  - GEMM2 (per tile): out(16x64) = weighted(16x960) x Wflat(960x64),
//    30 WMMAs/wave; B-operand pre-packed to per-lane fragment layout in f16
//    (one-time pack kernel) -> 2x16B vector loads per fragment from L2.
//  - GroupNorm stats span all 50k points -> two-pass with f32 atomics.
// ---------------------------------------------------------------------------

typedef __attribute__((ext_vector_type(16))) _Float16 v16h;
typedef __attribute__((ext_vector_type(8)))  _Float16 v8h;
typedef __attribute__((ext_vector_type(8)))  float    v8f;

#define N_S   50000
#define N_Q   50000
#define HN    32
#define KP    15
#define CIN   64
#define COUT  64
#define NG    8
#define SIGMA 0.6f
#define GEPS  1e-5f
#define NEG   0.1f
#define TM    16
#define NCHUNK (KP * 2)          // 30 K-chunks of 32 in GEMM2

// LDS layout (bytes)
#define W_OFF    0              // f16 [16][32][16]  -> 16384
#define F_OFF    16384          // f16 [16][32][64]  -> 65536
#define G_OFF    81920          // f16 [16][16][64]  -> 32768
#define IDX_OFF  114688         // int [512]         -> 2048
#define CNT_OFF  116736         // int [16]          -> 64
#define INV_OFF  116800         // f32 [16]          -> 64
#define GS_OFF   116864         // f32 [8]           -> 32
#define GQ_OFF   116896         // f32 [8]           -> 32
#define SMEM_BYTES 116928

// ---- one-time weight pre-pack into per-lane WMMA B-fragment layout ---------
// wpack[chunk][dcol][e2], e2 = half*16 + e in [0,32): 16 contiguous f16 per
// (lane,fragment) -> two 16B vector loads in GEMM2.
__launch_bounds__(256)
__global__ void pack_weights_kernel(const float* __restrict__ w,
                                    _Float16* __restrict__ wpack)
{
  const int i = blockIdx.x * 256 + threadIdx.x;
  if (i >= NCHUNK * COUT * 32) return;
  const int e2    = i & 31;
  const int dcol  = (i >> 5) & 63;
  const int chunk = i >> 11;
  const int kk = chunk >> 1;
  const int s  = chunk & 1;
  wpack[i] = (_Float16)w[((size_t)kk * CIN + s * 32 + e2) * COUT + dcol];
}

__launch_bounds__(128)
__global__ void kpconv_wmma_kernel(
    const float* __restrict__ s_feats,   // (N_S, 64)
    const float* __restrict__ q_points,  // (N_Q, 3)
    const float* __restrict__ s_points,  // (N_S, 3)
    const int*   __restrict__ nbr_idx,   // (N_Q, 32), shadow index == N_S
    const float* __restrict__ kpts,      // (15, 3)
    const _Float16* __restrict__ wpack,  // packed weights (B fragments)
    const float* __restrict__ bias,      // (64)
    float*       __restrict__ xws,       // (N_Q, 64) pre-GN output
    float*       __restrict__ gaccum)    // [8] sum, [8] sumsq
{
  extern __shared__ char smem[];
  _Float16* wLds = (_Float16*)(smem + W_OFF);
  _Float16* fLds = (_Float16*)(smem + F_OFF);
  _Float16* gLds = (_Float16*)(smem + G_OFF);
  int*      idxL = (int*)     (smem + IDX_OFF);
  int*      cntL = (int*)     (smem + CNT_OFF);
  float*    invN = (float*)   (smem + INV_OFF);
  float*    gsum = (float*)   (smem + GS_OFF);
  float*    gsq  = (float*)   (smem + GQ_OFF);

  const int tid  = threadIdx.x;       // 0..127
  const int lane = tid & 31;
  const int wave = tid >> 5;          // 0..3
  const int half = lane >> 4;         // 0 or 1
  const int l16  = lane & 15;
  const int tileBase = blockIdx.x * TM;

  if (tid < TM) cntL[tid] = 0;
  if (tid < NG) { gsum[tid] = 0.0f; gsq[tid] = 0.0f; }

  // ---- stage neighbor indices (coalesced) + prefetch gathered lines ----
  // Prefetches (global_prefetch_b8) put the random 256B feature rows and
  // 12B point rows in flight while we do the VALU-heavy distance phase.
  for (int i = tid; i < TM * HN; i += 128) {
    const int idx = nbr_idx[(size_t)tileBase * HN + i];
    idxL[i] = idx;
    if (idx < N_S) {
      const float* fp = s_feats + (size_t)idx * CIN;
      __builtin_prefetch(fp, 0, 0);
      __builtin_prefetch(fp + 32, 0, 0);           // second half of 256B row
      __builtin_prefetch(s_points + (size_t)idx * 3, 0, 0);
    }
  }
  __syncthreads();

  // ---- influence weights: w[m][h][k] = max(1 - |p_nbr - q - kp_k|/sigma, 0)
  for (int i = tid; i < TM * HN; i += 128) {
    const int m = i >> 5;
    const int idx = idxL[i];
    float px, py, pz;
    if (idx < N_S) {
      px = s_points[idx * 3 + 0];
      py = s_points[idx * 3 + 1];
      pz = s_points[idx * 3 + 2];
    } else {                      // shadow neighbor
      px = py = pz = 1e10f;
    }
    const float qx = q_points[(size_t)(tileBase + m) * 3 + 0];
    const float qy = q_points[(size_t)(tileBase + m) * 3 + 1];
    const float qz = q_points[(size_t)(tileBase + m) * 3 + 2];
    const float dx = px - qx, dy = py - qy, dz = pz - qz;
    _Float16* wr = wLds + (size_t)i * 16;
#pragma unroll
    for (int k = 0; k < KP; ++k) {
      const float ex = dx - kpts[k * 3 + 0];
      const float ey = dy - kpts[k * 3 + 1];
      const float ez = dz - kpts[k * 3 + 2];
      const float sq = ex * ex + ey * ey + ez * ez;
      const float wv = fmaxf(1.0f - sqrtf(sq) * (1.0f / SIGMA), 0.0f);
      wr[k] = (_Float16)wv;
    }
    wr[15] = (_Float16)0.0f;      // pad K=15 -> 16 with zero row
  }

  // ---- gather neighbor features (f32 -> f16, 16B LDS stores) + validity ----
  for (int i = tid; i < TM * HN; i += 128) {
    const int m = i >> 5;
    const int idx = idxL[i];
    _Float16* fr = fLds + (size_t)i * CIN;
    if (idx < N_S) {
      const float4* src = (const float4*)(s_feats + (size_t)idx * CIN);
      float sum = 0.0f;
#pragma unroll
      for (int j = 0; j < CIN / 8; ++j) {
        const float4 v0 = src[j * 2 + 0];
        const float4 v1 = src[j * 2 + 1];
        sum += v0.x + v0.y + v0.z + v0.w + v1.x + v1.y + v1.z + v1.w;
        v8h h8;
        h8[0] = (_Float16)v0.x; h8[1] = (_Float16)v0.y;
        h8[2] = (_Float16)v0.z; h8[3] = (_Float16)v0.w;
        h8[4] = (_Float16)v1.x; h8[5] = (_Float16)v1.y;
        h8[6] = (_Float16)v1.z; h8[7] = (_Float16)v1.w;
        *(v8h*)(fr + j * 8) = h8;
      }
      if (sum > 0.0f) atomicAdd(&cntL[m], 1);
    } else {
      const v8h z = {};
#pragma unroll
      for (int j = 0; j < CIN / 8; ++j) *(v8h*)(fr + j * 8) = z;
    }
  }
  __syncthreads();
  if (tid < TM) invN[tid] = 1.0f / fmaxf((float)cntL[tid], 1.0f);

  // ---- GEMM1: weighted_m(16x64) = w_m^T(16x32) x F_m(32x64) ----
  for (int mq = wave; mq < TM; mq += 4) {
    // A fragment: row = kernel point (l16), cols = h per CDNA5 16-bit A layout
    v16h a;
#pragma unroll
    for (int e = 0; e < 16; ++e) {
      const int h = ((e < 8) ? e : e + 8) + half * 8;
      a[e] = wLds[((size_t)mq * HN + h) * 16 + l16];
    }
#pragma unroll
    for (int nt = 0; nt < 4; ++nt) {
      // B fragment: col = channel (nt*16 + l16), rows = h (same K order as A)
      v16h b;
#pragma unroll
      for (int e = 0; e < 16; ++e) {
        const int h = ((e < 8) ? e : e + 8) + half * 8;
        b[e] = fLds[((size_t)mq * HN + h) * CIN + nt * 16 + l16];
      }
      v8f c = {};
      c = __builtin_amdgcn_wmma_f32_16x16x32_f16(
          false, a, false, b, (short)0, c, false, false);
#pragma unroll
      for (int r = 0; r < 8; ++r)
        gLds[((size_t)mq * 16 + (r + half * 8)) * CIN + nt * 16 + l16] =
            (_Float16)c[r];
    }
  }
  __syncthreads();

  // ---- GEMM2: out(16x64) = weighted(16x(15*64)) x Wflat((15*64)x64) ----
  // Each wave owns a 16-wide Cout tile; B fragments come pre-packed from
  // global (L2-resident), 2 x 16B vector loads each.
  const int dcol = wave * 16 + l16;
  v8f acc = {};
  for (int chunk = 0; chunk < NCHUNK; ++chunk) {
    const int kk = chunk >> 1;
    const int s  = chunk & 1;
    v16h a, b;
#pragma unroll
    for (int e = 0; e < 16; ++e) {
      const int kd = half * 16 + e;   // 0..31 in K-chunk
      a[e] = gLds[((size_t)l16 * 16 + kk) * CIN + s * 32 + kd];
    }
    const _Float16* wb = wpack + ((size_t)chunk * COUT + dcol) * 32 + half * 16;
    const v8h b0 = *(const v8h*)(wb);
    const v8h b1 = *(const v8h*)(wb + 8);
#pragma unroll
    for (int e = 0; e < 8; ++e) { b[e] = b0[e]; b[8 + e] = b1[e]; }
    acc = __builtin_amdgcn_wmma_f32_16x16x32_f16(
        false, a, false, b, (short)0, acc, false, false);
  }

  // ---- epilogue: neighbor-count normalize, bias, store + GN partials ----
  const float bc = bias[dcol];
  float ls = 0.0f, lq = 0.0f;
#pragma unroll
  for (int r = 0; r < 8; ++r) {
    const int mrow = r + half * 8;
    const float v = acc[r] * invN[mrow] + bc;
    xws[(size_t)(tileBase + mrow) * COUT + dcol] = v;
    ls += v;
    lq += v * v;
  }
  atomicAdd(&gsum[dcol >> 3], ls);   // group = channel / 8
  atomicAdd(&gsq[dcol >> 3], lq);
  __syncthreads();
  if (tid < NG) {
    atomicAdd(&gaccum[tid], gsum[tid]);
    atomicAdd(&gaccum[NG + tid], gsq[tid]);
  }
}

// ---- pass 2: GroupNorm (global stats) + affine + LeakyReLU -----------------
__launch_bounds__(256)
__global__ void gn_lrelu_kernel(const float* __restrict__ xws,
                                const float* __restrict__ gaccum,
                                const float* __restrict__ gamma,
                                const float* __restrict__ beta,
                                float* __restrict__ out)
{
  const int i = blockIdx.x * blockDim.x + threadIdx.x;
  if (i >= N_Q * COUT) return;
  const int c = i & (COUT - 1);
  const int g = c >> 3;
  const float cnt  = (float)N_Q * (float)(COUT / NG);   // elements per group
  const float mean = gaccum[g] / cnt;
  const float var  = gaccum[NG + g] / cnt - mean * mean;
  const float xn   = (xws[i] - mean) * rsqrtf(var + GEPS);
  const float y    = xn * gamma[c] + beta[c];
  out[i] = (y >= 0.0f) ? y : NEG * y;
}

extern "C" void kernel_launch(void* const* d_in, const int* in_sizes, int n_in,
                              void* d_out, int out_size, void* d_ws, size_t ws_size,
                              hipStream_t stream) {
  const float* s_feats  = (const float*)d_in[0];
  const float* q_points = (const float*)d_in[1];
  const float* s_points = (const float*)d_in[2];
  const int*   nbr_idx  = (const int*)  d_in[3];
  const float* kpts     = (const float*)d_in[4];
  const float* weights  = (const float*)d_in[5];
  const float* bias     = (const float*)d_in[6];
  const float* gamma    = (const float*)d_in[7];
  const float* beta     = (const float*)d_in[8];
  float* out = (float*)d_out;

  // workspace layout
  float*    gaccum = (float*)d_ws;                            // 16 f32
  float*    xws    = (float*)((char*)d_ws + 256);             // N_Q*64 f32
  _Float16* wpack  = (_Float16*)((char*)d_ws + 256 +
                                 (size_t)N_Q * COUT * sizeof(float));

  // zero the GroupNorm accumulators every launch (graph-capture safe)
  hipMemsetAsync(d_ws, 0, 16 * sizeof(float), stream);

  // pack weights into WMMA B-fragment layout (f16), once per launch
  const int packN = NCHUNK * COUT * 32;
  pack_weights_kernel<<<(packN + 255) / 256, 256, 0, stream>>>(weights, wpack);

  dim3 grid1(N_Q / TM);                            // 3125 blocks
  kpconv_wmma_kernel<<<grid1, 128, SMEM_BYTES, stream>>>(
      s_feats, q_points, s_points, nbr_idx, kpts, wpack, bias, xws, gaccum);

  const int total = N_Q * COUT;
  gn_lrelu_kernel<<<(total + 255) / 256, 256, 0, stream>>>(
      xws, gaccum, gamma, beta, out);
}
